// Propagate_36704790511690
// MI455X (gfx1250) — matrix operand order repaired and checked by reading, not compiled, not run on hardware
//
#include <hip/hip_runtime.h>
#include <hip/hip_bf16.h>

#define BB 8
#define CC 64
#define NN 4096
#define KK 16
#define EE 64
#define EPSV 1e-5f

typedef __attribute__((ext_vector_type(16))) __bf16 v16bf;
typedef __attribute__((ext_vector_type(8)))  __bf16 v8bf;
typedef __attribute__((ext_vector_type(8)))  float  v8f;
typedef __attribute__((ext_vector_type(4)))  float  vf4;
typedef __attribute__((ext_vector_type(2)))  float  vf2;

__device__ __forceinline__ v16bf cvt16(const float* v) {
    v16bf f;
#pragma unroll
    for (int i = 0; i < 16; ++i) f[i] = (__bf16)v[i];
    return f;
}

// Data-splitting butterfly reduction of 32 per-lane values across the 16
// lanes of a half-wave (masks 1,2,4,8 stay inside the half). 30 shuffles
// total. All private-array indices are compile-time constants; the
// lane-dependent role is resolved with scalar selects (v_cndmask), not
// dynamic register indexing. Afterwards each lane holds results for local
// indices li = 2*bitrev4(l15) (+0, +1) in v[0] and v[1].
template <bool IS_MAX>
__device__ __forceinline__ void halfwave_reduce32(float* v, int l15) {
#pragma unroll
    for (int st = 0; st < 4; ++st) {
        const int half = 16 >> st;
        const bool b = (l15 >> st) & 1;
#pragma unroll
        for (int i = 0; i < half; ++i) {
            const float lo = v[i];
            const float hh = v[i + half];
            const float send = b ? lo : hh;
            const float keep = b ? hh : lo;
            const float recv = __shfl_xor(send, 1 << st, 32);
            v[i] = IS_MAX ? fmaxf(keep, recv) : (keep + recv);
        }
    }
}

// After halfwave_reduce32: first local index this lane owns (always even).
__device__ __forceinline__ int rev_base(int l15) {
    return ((l15 & 1) << 4) | (((l15 >> 1) & 1) << 3) |
           (((l15 >> 2) & 1) << 2) | (((l15 >> 3) & 1) << 1);
}
// local index (8*t + r) -> channel
__device__ __forceinline__ int li_to_ch(int li, int hi) {
    return 16 * (li >> 3) + 8 * hi + (li & 7);
}

// Preload A-fragments for a 64x64 row-major weight matrix.
// Tile t covers rows [16t,16t+16), k-step s covers cols [32s,32s+32).
// A layout (contiguous k, matching B): lane m=l&15, hi=l>>4; elems 0..15 = K 16*hi+0..15
// (within the 32-wide k-step).
__device__ __forceinline__ void load_wfrags(const float* __restrict__ W,
                                            int l15, int hi, v16bf af[4][2]) {
#pragma unroll
    for (int t = 0; t < 4; ++t) {
#pragma unroll
        for (int s = 0; s < 2; ++s) {
            const float* wr = W + (size_t)(16 * t + l15) * CC + 32 * s + 16 * hi;
            float tw[16];
#pragma unroll
            for (int q = 0; q < 2; ++q) {
                vf4 w0 = *(const vf4*)(wr + 4 * q);
                vf4 w1 = *(const vf4*)(wr + 8 + 4 * q);
#pragma unroll
                for (int i = 0; i < 4; ++i) { tw[4 * q + i] = w0[i]; tw[8 + 4 * q + i] = w1[i]; }
            }
            af[t][s] = cvt16(tw);
        }
    }
}

// Tree-reduce stats and atomically publish (4 atomics per lane).
__device__ __forceinline__ void publish_stats(float* sm, float* sq, int l15, int hi,
                                              float* __restrict__ stat) {
    halfwave_reduce32<false>(sm, l15);
    halfwave_reduce32<false>(sq, l15);
    const int li  = rev_base(l15);
    const int ch0 = li_to_ch(li, hi);       // li even -> ch1 == ch0 + 1
    atomicAdd(&stat[ch0],          sm[0]);
    atomicAdd(&stat[ch0 + 1],      sm[1]);
    atomicAdd(&stat[64 + ch0],     sq[0]);
    atomicAdd(&stat[64 + ch0 + 1], sq[1]);
}

// ---------------- Kernel 0: transpose x (B,C,N) -> xT (B,N,C) ----------------
__global__ void k_transpose(const float* __restrict__ x, float* __restrict__ xT) {
    int tid = blockIdx.x * blockDim.x + threadIdx.x;     // (b,n,c), c fastest
    if (tid >= BB * NN * CC) return;
    int c = tid & (CC - 1);
    int n = (tid / CC) & (NN - 1);
    int b = tid / (CC * NN);
    xT[tid] = x[((size_t)b * CC + c) * NN + n];
}

// ---------------- Kernel 1: edge features + conv1 (WMMA) + BN1 stats --------
__global__ void __launch_bounds__(256) k_layer1(const float* __restrict__ xT,
                                                const int* __restrict__ idx,
                                                const float* __restrict__ W1,
                                                __bf16* __restrict__ h1,
                                                float* __restrict__ stat) {
    const int lane = threadIdx.x & 31;
    const int l15  = lane & 15;
    const int hi   = lane >> 4;
    const int waveId = (blockIdx.x * blockDim.x + threadIdx.x) >> 5;
    const int nWaves = (gridDim.x * blockDim.x) >> 5;

    v16bf af[4][2];
    load_wfrags(W1, l15, hi, af);

    float sm[32], sq[32];
#pragma unroll
    for (int i = 0; i < 32; ++i) { sm[i] = 0.f; sq[i] = 0.f; }

    for (int pt = waveId; pt < BB * NN; pt += nWaves) {
        const int b = pt >> 12;                 // N = 4096
        const int j = idx[(size_t)pt * KK + l15];
        const float* nb = xT + ((size_t)b * NN + j) * CC;
        const float* ct = xT + (size_t)pt * CC;

        // B fragment: lane column = neighbor l15; elems = 16 contiguous channels
        // starting at 32s + 16*hi.
        v16bf bf[2];
#pragma unroll
        for (int s = 0; s < 2; ++s) {
            const int c0 = 32 * s + 16 * hi;
            float tmp[16];
#pragma unroll
            for (int q = 0; q < 4; ++q) {
                vf4 a = *(const vf4*)(nb + c0 + 4 * q);
                vf4 c = *(const vf4*)(ct + c0 + 4 * q);
#pragma unroll
                for (int i = 0; i < 4; ++i) tmp[4 * q + i] = a[i] - c[i];
            }
            bf[s] = cvt16(tmp);
        }

        v8f acc[4];
#pragma unroll
        for (int t = 0; t < 4; ++t) { v8f z = {}; acc[t] = z; }
#pragma unroll
        for (int s = 0; s < 2; ++s)
#pragma unroll
            for (int t = 0; t < 4; ++t)
                acc[t] = __builtin_amdgcn_wmma_f32_16x16x32_bf16(
                    false, af[t][s], false, bf[s], (short)0, acc[t], false, false);

        // stats + bf16 store; lane holds channels 16t + 8*hi + r, column l15
        __bf16* hp = h1 + ((size_t)pt * KK + l15) * EE;
#pragma unroll
        for (int t = 0; t < 4; ++t) {
            v8bf o;
#pragma unroll
            for (int r = 0; r < 8; ++r) {
                float v = acc[t][r];
                sm[8 * t + r] += v;
                sq[8 * t + r] = fmaf(v, v, sq[8 * t + r]);
                o[r] = (__bf16)v;
            }
            *(v8bf*)(hp + 16 * t + 8 * hi) = o;
        }
    }
    publish_stats(sm, sq, l15, hi, stat);
}

// ---------------- BN finalize: A = g*rsqrt(var+eps), C = beta - A*mean ------
__global__ void k_finalize(const float* __restrict__ stat, const float* __restrict__ g,
                           const float* __restrict__ be, float inv_cnt,
                           float* __restrict__ ac) {
    int ch = threadIdx.x;
    if (ch >= EE) return;
    float mean = stat[ch] * inv_cnt;
    float var  = stat[64 + ch] * inv_cnt - mean * mean;
    float A    = g[ch] * rsqrtf(var + EPSV);
    ac[ch]      = A;
    ac[64 + ch] = be[ch] - A * mean;
}

// ------- Kernel 2: BN1+ReLU (folded affine) + conv2 (WMMA) + stats + max_k --
__global__ void __launch_bounds__(256) k_layer2(const __bf16* __restrict__ h1,
                                                const float* __restrict__ W2,
                                                const float* __restrict__ ac1,
                                                float* __restrict__ m2,
                                                float* __restrict__ stat) {
    const int lane = threadIdx.x & 31;
    const int l15  = lane & 15;
    const int hi   = lane >> 4;
    const int waveId = (blockIdx.x * blockDim.x + threadIdx.x) >> 5;
    const int nWaves = (gridDim.x * blockDim.x) >> 5;

    v16bf af[4][2];
    load_wfrags(W2, l15, hi, af);

    float av[2][16], cv[2][16];
#pragma unroll
    for (int s = 0; s < 2; ++s)
#pragma unroll
        for (int i = 0; i < 16; ++i) {
            int ch = 32 * s + 16 * hi + i;
            av[s][i] = ac1[ch];
            cv[s][i] = ac1[64 + ch];
        }

    float sm[32], sq[32];
#pragma unroll
    for (int i = 0; i < 32; ++i) { sm[i] = 0.f; sq[i] = 0.f; }

    const int liBase = rev_base(l15);
    const int chBase = li_to_ch(liBase, hi);

    for (int pt = waveId; pt < BB * NN; pt += nWaves) {
        const __bf16* hp = h1 + ((size_t)pt * KK + l15) * EE;

        v16bf bf[2];
#pragma unroll
        for (int s = 0; s < 2; ++s) {
            const __bf16* p = hp + 32 * s + 16 * hi;
            v8bf x0 = *(const v8bf*)(p);
            v8bf x1 = *(const v8bf*)(p + 8);
            v16bf f;
#pragma unroll
            for (int i = 0; i < 8; ++i) {
                f[i]     = (__bf16)fmaxf((float)x0[i] * av[s][i]     + cv[s][i],     0.f);
                f[8 + i] = (__bf16)fmaxf((float)x1[i] * av[s][8 + i] + cv[s][8 + i], 0.f);
            }
            bf[s] = f;
        }

        v8f acc[4];
#pragma unroll
        for (int t = 0; t < 4; ++t) { v8f z = {}; acc[t] = z; }
#pragma unroll
        for (int s = 0; s < 2; ++s)
#pragma unroll
            for (int t = 0; t < 4; ++t)
                acc[t] = __builtin_amdgcn_wmma_f32_16x16x32_bf16(
                    false, af[t][s], false, bf[s], (short)0, acc[t], false, false);

        // stats over all K (pre-max), then tree-max over the 16 neighbor
        // columns: max_k relu(a*h+c) == relu(a*max_k h + c) for a > 0.
        float mv[32];
#pragma unroll
        for (int t = 0; t < 4; ++t)
#pragma unroll
            for (int r = 0; r < 8; ++r) {
                float v = acc[t][r];
                sm[8 * t + r] += v;
                sq[8 * t + r] = fmaf(v, v, sq[8 * t + r]);
                mv[8 * t + r] = v;
            }
        halfwave_reduce32<true>(mv, l15);
        // every lane stores 2 adjacent channels' maxima (8B aligned store)
        vf2 st2 = {mv[0], mv[1]};
        *(vf2*)(m2 + (size_t)pt * EE + chBase) = st2;
    }
    publish_stats(sm, sq, l15, hi, stat);
}

// ------- Kernel 3: BN2+ReLU + conv3 (WMMA, 16 points per wave) + stats ------
__global__ void __launch_bounds__(256) k_layer3(const float* __restrict__ m2,
                                                const float* __restrict__ W3,
                                                const float* __restrict__ ac2,
                                                float* __restrict__ h3,
                                                float* __restrict__ stat) {
    const int lane = threadIdx.x & 31;
    const int l15  = lane & 15;
    const int hi   = lane >> 4;
    const int waveId = (blockIdx.x * blockDim.x + threadIdx.x) >> 5;
    const int nWaves = (gridDim.x * blockDim.x) >> 5;

    v16bf af[4][2];
    load_wfrags(W3, l15, hi, af);

    float av[2][16], cv[2][16];
#pragma unroll
    for (int s = 0; s < 2; ++s)
#pragma unroll
        for (int i = 0; i < 16; ++i) {
            int ch = 32 * s + 16 * hi + i;
            av[s][i] = ac2[ch];
            cv[s][i] = ac2[64 + ch];
        }

    float sm[32], sq[32];
#pragma unroll
    for (int i = 0; i < 32; ++i) { sm[i] = 0.f; sq[i] = 0.f; }

    const int nGroups = (BB * NN) / 16;
    for (int grp = waveId; grp < nGroups; grp += nWaves) {
        const int p = grp * 16 + l15;           // this lane's point/column
        const float* mp = m2 + (size_t)p * EE;

        v16bf bf[2];
#pragma unroll
        for (int s = 0; s < 2; ++s) {
            const int c0 = 32 * s + 16 * hi;
            float tmp[16];
#pragma unroll
            for (int q = 0; q < 4; ++q) {
                vf4 a = *(const vf4*)(mp + c0 + 4 * q);
#pragma unroll
                for (int i = 0; i < 4; ++i) {
                    int k = 4 * q + i;
                    tmp[k] = fmaxf(fmaf(a[i], av[s][k], cv[s][k]), 0.f);
                }
            }
            bf[s] = cvt16(tmp);
        }

        v8f acc[4];
#pragma unroll
        for (int t = 0; t < 4; ++t) { v8f z = {}; acc[t] = z; }
#pragma unroll
        for (int s = 0; s < 2; ++s)
#pragma unroll
            for (int t = 0; t < 4; ++t)
                acc[t] = __builtin_amdgcn_wmma_f32_16x16x32_bf16(
                    false, af[t][s], false, bf[s], (short)0, acc[t], false, false);

        float* op = h3 + (size_t)p * EE;
#pragma unroll
        for (int t = 0; t < 4; ++t) {
            vf4 s0, s1;
#pragma unroll
            for (int r = 0; r < 8; ++r) {
                float v = acc[t][r];
                sm[8 * t + r] += v;
                sq[8 * t + r] = fmaf(v, v, sq[8 * t + r]);
                if (r < 4) s0[r] = v; else s1[r - 4] = v;
            }
            *(vf4*)(op + 16 * t + 8 * hi)     = s0;
            *(vf4*)(op + 16 * t + 8 * hi + 4) = s1;
        }
    }
    publish_stats(sm, sq, l15, hi, stat);
}

// ---------------- Kernel 4: apply BN3 + ReLU, write (B,E,N) output ----------
__global__ void k_output(const float* __restrict__ h3, const float* __restrict__ ac3,
                         float* __restrict__ out) {
    int tid = blockIdx.x * blockDim.x + threadIdx.x;    // (b,e,n), n fastest
    if (tid >= BB * EE * NN) return;
    int n = tid & (NN - 1);
    int e = (tid >> 12) & (EE - 1);
    int b = tid >> 18;
    float v = h3[((size_t)b * NN + n) * EE + e];
    out[tid] = fmaxf(fmaf(ac3[e], v, ac3[64 + e]), 0.f);
}

extern "C" void kernel_launch(void* const* d_in, const int* in_sizes, int n_in,
                              void* d_out, int out_size, void* d_ws, size_t ws_size,
                              hipStream_t stream) {
    const float* x   = (const float*)d_in[0];
    const int*   idx = (const int*)d_in[1];
    const float* W1  = (const float*)d_in[2];
    const float* g1  = (const float*)d_in[4];
    const float* be1 = (const float*)d_in[5];
    const float* W2  = (const float*)d_in[6];
    const float* g2  = (const float*)d_in[8];
    const float* be2 = (const float*)d_in[9];
    const float* W3  = (const float*)d_in[10];
    const float* g3  = (const float*)d_in[12];
    const float* be3 = (const float*)d_in[13];
    float* out = (float*)d_out;

    // Workspace layout
    char* ws = (char*)d_ws;
    const size_t xT_bytes = (size_t)BB * NN * CC * sizeof(float);        // 8 MB
    const size_t h1_bytes = (size_t)BB * NN * KK * EE * sizeof(__bf16);  // 64 MB
    const size_t m2_bytes = (size_t)BB * NN * EE * sizeof(float);        // 8 MB
    const size_t h3_bytes = (size_t)BB * NN * EE * sizeof(float);        // 8 MB
    float*  xT  = (float*)ws;
    __bf16* h1  = (__bf16*)(ws + xT_bytes);
    float*  m2  = (float*)(ws + xT_bytes + h1_bytes);
    float*  h3  = (float*)(ws + xT_bytes + h1_bytes + m2_bytes);
    float*  st  = (float*)(ws + xT_bytes + h1_bytes + m2_bytes + h3_bytes);
    float* st1 = st;        float* st2 = st + 128;  float* st3 = st + 256;
    float* ac1 = st + 384;  float* ac2 = st + 512;  float* ac3 = st + 640;

    hipMemsetAsync(st, 0, 384 * sizeof(float), stream);

    k_transpose<<<(BB * NN * CC) / 256, 256, 0, stream>>>(x, xT);
    k_layer1<<<1024, 256, 0, stream>>>(xT, idx, W1, h1, st1);
    k_finalize<<<1, 64, 0, stream>>>(st1, g1, be1, 1.f / (float)(BB * NN * KK), ac1);
    k_layer2<<<1024, 256, 0, stream>>>(h1, W2, ac1, m2, st2);
    k_finalize<<<1, 64, 0, stream>>>(st2, g2, be2, 1.f / (float)(BB * NN * KK), ac2);
    k_layer3<<<256, 256, 0, stream>>>(m2, W3, ac2, h3, st3);
    k_finalize<<<1, 64, 0, stream>>>(st3, g3, be3, 1.f / (float)(BB * NN), ac3);
    k_output<<<(BB * EE * NN) / 256, 256, 0, stream>>>(h3, ac3, out);
}
